// Head_6777458393226
// MI455X (gfx1250) — matrix operand order repaired and checked by reading, not compiled, not run on hardware
//
#include <hip/hip_runtime.h>
#include <hip/hip_bf16.h>

// CDNA5 / gfx1250 flash-attention head, bf16 WMMA (v_wmma_f32_16x16x32_bf16),
// b128-wide fragment movement, exp2-domain online softmax (raw v_exp_f32).
typedef __attribute__((ext_vector_type(16))) __bf16 v16bf;
typedef __attribute__((ext_vector_type(8)))  __bf16 v8bf;
typedef __attribute__((ext_vector_type(8)))  float  v8f;
typedef __attribute__((ext_vector_type(4)))  float  v4f;

constexpr int  Bb = 8, Ts = 2048, Cc = 1024, Hh = 64;
// C^-0.5 * log2(e): softmax runs in the exp2 domain so every exponential is a
// single v_exp_f32 and the attention scale costs zero extra VALU per element.
constexpr float SCALE2 = 0.03125f * 1.4426950408889634f;
constexpr int  QW = 4;              // query-tile waves per attention block

__device__ __forceinline__ v8f wmma_bf16(const v16bf& a, const v16bf& b, v8f c) {
  // 8 args: (neg_a, A, neg_b, B, c_mod, C, reuse_a, reuse_b)
  return __builtin_amdgcn_wmma_f32_16x16x32_bf16(false, a, false, b, (short)0, c,
                                                 false, false);
}

__device__ __forceinline__ v16bf join8(v8bf lo, v8bf hi) {
  v16bf r;
#pragma unroll
  for (int e = 0; e < 8; ++e) { r[e] = lo[e]; r[e + 8] = hi[e]; }
  return r;
}

// ---------------------------------------------------------------------------
// Kernel 0: one-shot W transpose+downconvert: W[C,H] f32 -> Wt[H,C] bf16 (x3).
// Makes the projection B-operand a single contiguous v16bf load per WMMA.
// ---------------------------------------------------------------------------
__global__ __launch_bounds__(256) void wt_kernel(
    const float* __restrict__ Wq, const float* __restrict__ Wk,
    const float* __restrict__ Wv, __bf16* __restrict__ wT)
{
  const int idx = blockIdx.x * 256 + threadIdx.x;   // [3][64][1024]
  const int mat = idx >> 16;
  const int rem = idx & 65535;
  const int h = rem >> 10;
  const int c = rem & 1023;
  const float* W = (mat == 0) ? Wq : (mat == 1) ? Wk : Wv;
  wT[idx] = (__bf16)W[c * Hh + h];
}

// ---------------------------------------------------------------------------
// Kernel 1: q/k/v projection.  x[B,T,C] f32 x Wt[H,C] bf16 -> q,k bf16 [B,T,H]
// and vT bf16 [B,H,T] (transposed so attention's P*V B-operand is contiguous).
// Block = 12 waves: wave -> {q,k,v} x {16-col H tile}; 32 WMMAs per wave.
// ---------------------------------------------------------------------------
__global__ __launch_bounds__(384) void qkv_proj_kernel(
    const float* __restrict__ x, const __bf16* __restrict__ wT,
    __bf16* __restrict__ qb, __bf16* __restrict__ kb, __bf16* __restrict__ vT)
{
  const int lane = threadIdx.x & 31;
  const int wid  = threadIdx.x >> 5;      // 0..11
  const int mat  = wid >> 2;              // 0=q 1=k 2=v
  const int ht   = wid & 3;               // 16-wide H tile
  const int b    = blockIdx.y;
  const int t0   = blockIdx.x * 16;
  const int row  = lane & 15;             // A: lane%16 = M row
  const int hi   = lane >> 4;             // lane half -> K sub-range
  const int n    = lane & 15;             // B/C: lane%16 = N col

  const float*  xrow = x + ((size_t)b * Ts + (t0 + row)) * Cc;
  const __bf16* wcol = wT + (size_t)mat * (Hh * Cc) + (size_t)(ht * 16 + n) * Cc;

  v8f acc = {};
  for (int kc = 0; kc < Cc; kc += 32) {
    // A 16x32 bf16: elems 0-15 -> K=hi*16+0..15
    const float* xp = xrow + kc + hi * 16;
    const v4f x0 = *(const v4f*)(xp);
    const v4f x1 = *(const v4f*)(xp + 4);
    const v4f x2 = *(const v4f*)(xp + 8);
    const v4f x3 = *(const v4f*)(xp + 12);
    v16bf a;
#pragma unroll
    for (int e = 0; e < 4; ++e) {
      a[e]      = (__bf16)x0[e];
      a[e + 4]  = (__bf16)x1[e];
      a[e + 8]  = (__bf16)x2[e];
      a[e + 12] = (__bf16)x3[e];
    }
    // B 32x16 bf16: lane half hi covers K = hi*16 + e (contiguous in Wt row)
    const v16bf bf = *(const v16bf*)(wcol + kc + hi * 16);
    acc = wmma_bf16(a, bf, acc);
  }

  // C layout: VGPR r -> row (r + 8*hi), col n
  if (mat == 2) {
    // vT[b][h][t]: rows r are t-contiguous -> one packed 16B store per lane
    v8bf pk;
#pragma unroll
    for (int r = 0; r < 8; ++r) pk[r] = (__bf16)acc[r];
    *(v8bf*)(vT + ((size_t)b * Hh + ht * 16 + n) * Ts + t0 + hi * 8) = pk;
  } else {
    __bf16* outm = (mat == 0) ? qb : kb;
#pragma unroll
    for (int r = 0; r < 8; ++r) {
      const int m = r + hi * 8;
      outm[((size_t)b * Ts + (t0 + m)) * Hh + ht * 16 + n] = (__bf16)acc[r];
    }
  }
}

// ---------------------------------------------------------------------------
// Kernel 2: fused causal flash attention.  One wave per 16-query tile; online
// softmax (exp2 domain) over kv-tiles of 32.  Per iter: 4 WMMA (S = Q K^T),
// softmax with 16-lane shfl_xor reductions, P relayout C->A via per-wave LDS
// (s_wait_dscnt fence, b128 DS reloads), 4 WMMA (O += P V).
// ---------------------------------------------------------------------------
__global__ __launch_bounds__(32 * QW) void flash_attn_kernel(
    const __bf16* __restrict__ qb, const __bf16* __restrict__ kb,
    const __bf16* __restrict__ vT, float* __restrict__ out)
{
  __shared__ __bf16 pbuf[QW][16][32];

  const int lane = threadIdx.x & 31;
  const int wid  = threadIdx.x >> 5;
  const int b    = blockIdx.y;
  const int t0   = (blockIdx.x * QW + wid) * 16;
  const int row  = lane & 15;
  const int hi   = lane >> 4;
  const int n    = lane & 15;

  // Q A-fragments: two 16x32 slabs over H=64, each = two 16B loads
  const __bf16* qr = qb + ((size_t)b * Ts + (t0 + row)) * Hh;
  const v16bf aq0 = join8(*(const v8bf*)(qr + hi * 16),
                          *(const v8bf*)(qr + hi * 16 + 8));
  const v16bf aq1 = join8(*(const v8bf*)(qr + 32 + hi * 16),
                          *(const v8bf*)(qr + 32 + hi * 16 + 8));

  v8f o0 = {}, o1 = {}, o2 = {}, o3 = {};
  float m_i[8], l_i[8];                      // running row max / sum, log2-e dom.
#pragma unroll
  for (int r = 0; r < 8; ++r) { m_i[r] = -1e30f; l_i[r] = 0.0f; }

  const int jmax = (t0 + 16 + 31) / 32;
  for (int j = 0; j < jmax; ++j) {
    const int kv0 = j * 32;

    // K^T B-fragments: per-lane column = kv row of K, K-dim contiguous in H.
    const __bf16* krow0 = kb + ((size_t)b * Ts + (kv0 + n)) * Hh;       // st=0
    const __bf16* krow1 = kb + ((size_t)b * Ts + (kv0 + 16 + n)) * Hh;  // st=1
    const v16bf bk00 = *(const v16bf*)(krow0 + hi * 16);
    const v16bf bk10 = *(const v16bf*)(krow0 + 32 + hi * 16);
    const v16bf bk01 = *(const v16bf*)(krow1 + hi * 16);
    const v16bf bk11 = *(const v16bf*)(krow1 + 32 + hi * 16);

    v8f s0 = {}, s1 = {};
    s0 = wmma_bf16(aq0, bk00, s0);  s0 = wmma_bf16(aq1, bk10, s0);
    s1 = wmma_bf16(aq0, bk01, s1);  s1 = wmma_bf16(aq1, bk11, s1);

    if (j + 1 < jmax) {  // next K/V tile toward caches (global_prefetch_b8)
      __builtin_prefetch(&kb[((size_t)b * Ts + (kv0 + 32 + n)) * Hh], 0, 1);
      __builtin_prefetch(&vT[((size_t)b * Hh + n) * Ts + kv0 + 32], 0, 1);
    }

    // online softmax in exp2 domain, per C-register row (row = r + 8*hi)
#pragma unroll
    for (int r = 0; r < 8; ++r) {
      const int grow = t0 + r + hi * 8;
      float sv0 = s0[r] * SCALE2;            // scale & log2e folded together
      float sv1 = s1[r] * SCALE2;
      if (kv0 + n > grow)      sv0 = -1e30f; // causal mask
      if (kv0 + 16 + n > grow) sv1 = -1e30f;
      float rmax = fmaxf(sv0, sv1);
#pragma unroll
      for (int off = 1; off < 16; off <<= 1)
        rmax = fmaxf(rmax, __shfl_xor(rmax, off, 16));
      const float mnew = fmaxf(m_i[r], rmax);
      const float corr = __builtin_amdgcn_exp2f(m_i[r] - mnew);  // v_exp_f32
      const float p0   = __builtin_amdgcn_exp2f(sv0 - mnew);
      const float p1   = __builtin_amdgcn_exp2f(sv1 - mnew);
      float rsum = p0 + p1;
#pragma unroll
      for (int off = 1; off < 16; off <<= 1)
        rsum += __shfl_xor(rsum, off, 16);
      l_i[r] = l_i[r] * corr + rsum;
      m_i[r] = mnew;
      o0[r] *= corr;  o1[r] *= corr;  o2[r] *= corr;  o3[r] *= corr;
      // stage P (C layout) into LDS for the A-layout reload
      pbuf[wid][r + hi * 8][n]      = (__bf16)p0;
      pbuf[wid][r + hi * 8][16 + n] = (__bf16)p1;
    }
    // per-wave DS fence: DS ops are in-order per wave; this stops the
    // compiler from hoisting the reload above the stores.
    asm volatile("s_wait_dscnt 0" ::: "memory");

    // reload P as 16x32 bf16 A-fragment: two ds_load_b128 per lane
    const __bf16* prow = &pbuf[wid][row][0];
    const v16bf pa = join8(*(const v8bf*)(prow + hi * 16),
                           *(const v8bf*)(prow + hi * 16 + 8));

    // V B-fragments from vT[b][h][t]: K-dim (kv) contiguous -> one v16bf each
    const __bf16* vt0 = vT + ((size_t)b * Hh + n) * Ts + kv0 + hi * 16;
    const v16bf bv0 = *(const v16bf*)(vt0);
    const v16bf bv1 = *(const v16bf*)(vt0 + (size_t)16 * Ts);
    const v16bf bv2 = *(const v16bf*)(vt0 + (size_t)32 * Ts);
    const v16bf bv3 = *(const v16bf*)(vt0 + (size_t)48 * Ts);

    o0 = wmma_bf16(pa, bv0, o0);
    o1 = wmma_bf16(pa, bv1, o1);
    o2 = wmma_bf16(pa, bv2, o2);
    o3 = wmma_bf16(pa, bv3, o3);
  }

  // epilogue: normalize by row sums, store f32
#pragma unroll
  for (int r = 0; r < 8; ++r) {
    const int m = r + hi * 8;
    const float inv = 1.0f / l_i[r];
    float* orow = out + ((size_t)b * Ts + (t0 + m)) * Hh + n;
    orow[0]  = o0[r] * inv;
    orow[16] = o1[r] * inv;
    orow[32] = o2[r] * inv;
    orow[48] = o3[r] * inv;
  }
}

// ---------------------------------------------------------------------------
extern "C" void kernel_launch(void* const* d_in, const int* in_sizes, int n_in,
                              void* d_out, int out_size, void* d_ws, size_t ws_size,
                              hipStream_t stream) {
  const float* x  = (const float*)d_in[0];
  const float* Wq = (const float*)d_in[1];
  const float* Wk = (const float*)d_in[2];
  const float* Wv = (const float*)d_in[3];
  float* out = (float*)d_out;

  const size_t nqkv = (size_t)Bb * Ts * Hh;     // 1,048,576 elements each
  __bf16* qb = (__bf16*)d_ws;                   // [B,T,H]
  __bf16* kb = qb + nqkv;                       // [B,T,H]
  __bf16* vT = kb + nqkv;                       // [B,H,T]
  __bf16* wT = vT + nqkv;                       // [3,H,C] = 384 KB

  wt_kernel<<<dim3(3 * Hh * Cc / 256), 256, 0, stream>>>(Wq, Wk, Wv, wT);

  dim3 pgrid(Ts / 16, Bb);                      // (128, 8)
  qkv_proj_kernel<<<pgrid, 384, 0, stream>>>(x, wT, qb, kb, vT);

  dim3 agrid(Ts / (16 * QW), Bb);               // (32, 8)
  flash_attn_kernel<<<agrid, 32 * QW, 0, stream>>>(qb, kb, vT, out);
}
